// MultiScaleRetention_60258391162953
// MI455X (gfx1250) — matrix operand order repaired and checked by reading, not compiled, not run on hardware
//
#include <hip/hip_runtime.h>
#include <stdint.h>

// ---------------------------------------------------------------------------
// MultiScaleRetention forward for gfx1250 (MI455X), bf16 WMMA pipeline.
// B=2, S=2048, HIDDEN=1024, HEADS=8, HEAD=128.
// ---------------------------------------------------------------------------

#define Bsz   2
#define Ssz   2048
#define HID   1024
#define NH    8
#define HD    128
#define ROWS  (Bsz * Ssz)          // 4096 total token rows

typedef __attribute__((ext_vector_type(16))) __bf16   v16bf;
typedef __attribute__((ext_vector_type(8)))  float    v8f;
typedef __attribute__((ext_vector_type(4)))  unsigned int u32x4;

typedef unsigned short ushort_t;

__device__ __forceinline__ ushort_t f2bf(float f) {
    union { float f; unsigned u; } a; a.f = f;
    unsigned u = a.u;
    u += 0x7fffu + ((u >> 16) & 1u);          // round-to-nearest-even
    return (ushort_t)(u >> 16);
}

union V16 { v16bf v; u32x4 q[2]; };

// A-operand (16xK row-major, 16-bit): lane(r,hi) holds row r, K = {kc+hi*8+0..7, kc+16+hi*8+0..7}
__device__ __forceinline__ v16bf load_a16(const ushort_t* row, int kc, int hi) {
    V16 r;
    r.q[0] = *(const u32x4*)(row + kc + hi * 8);
    r.q[1] = *(const u32x4*)(row + kc + 16 + hi * 8);
    return r.v;
}
// B-operand (Kx16, column c per lane, read from pre-transposed BT row c): K = {kc+hi*8+0..7, kc+16+hi*8+0..7}
__device__ __forceinline__ v16bf load_b16(const ushort_t* btrow, int kc, int hi) {
    V16 r;
    const ushort_t* p = btrow + kc + hi * 8;
    r.q[0] = *(const u32x4*)(p);
    r.q[1] = *(const u32x4*)(p + 16);
    return r.v;
}
__device__ __forceinline__ v8f wmma_bf(v16bf a, v16bf b, v8f c) {
    return __builtin_amdgcn_wmma_f32_16x16x32_bf16(false, a, false, b, (short)0, c, false, false);
}

// ---------------------------------------------------------------------------
// Conversion / transpose kernels (fp32 -> bf16)
// ---------------------------------------------------------------------------
__global__ __launch_bounds__(256) void cvt_bf16_kernel(const float* __restrict__ in,
                                                       ushort_t* __restrict__ out, int n) {
    int i = blockIdx.x * 256 + threadIdx.x;
    if (i < n) out[i] = f2bf(in[i]);
}

// [H,128,128] -> transposed per head: out[h][e][d] = in[h][d][e]
__global__ __launch_bounds__(256) void transpose_head_kernel(const float* __restrict__ in,
                                                             ushort_t* __restrict__ out) {
    int i = blockIdx.x * 256 + threadIdx.x;           // n = 8*128*128 = 131072
    if (i >= NH * HD * HD) return;
    int h = i >> 14, r = (i >> 7) & 127, c = i & 127;
    out[(h << 14) + c * HD + r] = f2bf(in[i]);
}

// [1024,1024] -> transposed: out[c][r] = in[r][c]
__global__ __launch_bounds__(256) void transpose_1024_kernel(const float* __restrict__ in,
                                                             ushort_t* __restrict__ out) {
    int i = blockIdx.x * 256 + threadIdx.x;           // n = 1048576
    if (i >= HID * HID) return;
    int r = i >> 10, c = i & 1023;
    out[c * HID + r] = f2bf(in[i]);
}

// ---------------------------------------------------------------------------
// K1: per-head QKV projections.  One wave = one full 16x128 row block for one
// of q/k/v (8 accumulator tiles, K=128).  Per K-chunk: stage A + all 8 B
// operands, then fire 8 back-to-back WMMAs (one loadcnt wait per chunk).
//   q,k stored [B,H,S,128] row-major bf16 ; v stored transposed [B,H,128,S]
//   (vT store is fully packed: one b128 per lane per tile).
// ---------------------------------------------------------------------------
__global__ __launch_bounds__(256) void qkv_kernel(const ushort_t* __restrict__ xbf,
                                                  const ushort_t* __restrict__ wT,  // WqT|WkT|WvT
                                                  ushort_t* __restrict__ qbf,
                                                  ushort_t* __restrict__ kbf,
                                                  ushort_t* __restrict__ vTbf) {
    const int lane = threadIdx.x & 31;
    const int wid  = threadIdx.x >> 5;
    int t = blockIdx.x * 8 + wid;                     // 3 * 2*8*128 = 6144 tasks
    const int which = t / 2048;   t %= 2048;
    const int b  = t / 1024;      t %= 1024;
    const int h  = t / 128;
    const int nb = t % 128;
    const int n0 = nb * 16;
    const int c  = lane & 15, hi = lane >> 4;

    const ushort_t* arow   = xbf + (size_t)(b * Ssz + n0 + c) * HID + h * HD;  // A row = lane&15
    const ushort_t* wtbase = wT + (size_t)which * (NH * HD * HD) + h * (HD * HD);

    v8f acc[8];
#pragma unroll
    for (int dt = 0; dt < 8; ++dt) acc[dt] = (v8f){};

#pragma unroll
    for (int j = 0; j < 4; ++j) {
        const int kc = 32 * j;
        const v16bf a = load_a16(arow, kc, hi);       // reused by 8 WMMAs
        v16bf bb[8];
#pragma unroll
        for (int dt = 0; dt < 8; ++dt)
            bb[dt] = load_b16(wtbase + (dt * 16 + c) * HD, kc, hi);
#pragma unroll
        for (int dt = 0; dt < 8; ++dt)
            acc[dt] = wmma_bf(a, bb[dt], acc[dt]);
    }

    if (which == 2) {
        // vT [B,H,128,S]: lane's 8 rows are contiguous along S -> one b128 store per tile
        ushort_t* vb = vTbf + (size_t)(b * NH + h) * HD * Ssz;
#pragma unroll
        for (int dt = 0; dt < 8; ++dt) {
            const int col = dt * 16 + c;
            u32x4 pk;
#pragma unroll
            for (int g = 0; g < 4; ++g)
                pk[g] = (unsigned)f2bf(acc[dt][2 * g]) | ((unsigned)f2bf(acc[dt][2 * g + 1]) << 16);
            *(u32x4*)(vb + (size_t)col * Ssz + n0 + 8 * hi) = pk;
        }
    } else {
        ushort_t* ob = (which ? kbf : qbf) + (size_t)(b * NH + h) * Ssz * HD;
#pragma unroll
        for (int dt = 0; dt < 8; ++dt) {
            const int col = dt * 16 + c;
#pragma unroll
            for (int g = 0; g < 8; ++g)
                ob[(size_t)(n0 + g + 8 * hi) * HD + col] = f2bf(acc[dt][g]);
        }
    }
}

// ---------------------------------------------------------------------------
// K2: chunked causal retention + decay + GroupNorm.
//   One wave owns 16 query rows (n0..n0+15) and all d=128 output cols.
//   Loops over K=32 key chunks: scores via WMMA, decay mask in fp32,
//   bf16 score tile (lane transpose via LDS) feeds the scores@v WMMA.
//   B-operands staged per chunk so WMMAs issue back-to-back.
// ---------------------------------------------------------------------------
__global__ __launch_bounds__(256) void attn_kernel(const ushort_t* __restrict__ qbf,
                                                   const ushort_t* __restrict__ kbf,
                                                   const ushort_t* __restrict__ vTbf,
                                                   const float* __restrict__ gnw,
                                                   const float* __restrict__ gnb,
                                                   float* __restrict__ ret) {
    __shared__ __attribute__((aligned(16))) ushort_t sc[8][16][32];

    const int lane = threadIdx.x & 31;
    const int wid  = threadIdx.x >> 5;
    int t = blockIdx.x * 8 + wid;                     // 2*8*128 = 2048 tasks
    const int b  = t / (NH * 128); t %= (NH * 128);
    const int h  = t / 128;
    const int nb = t % 128;
    const int n0 = nb * 16;
    const int c  = lane & 15, hi = lane >> 4;

    // gamma for this head: lin = linspace(log(1/512), log(1/32), 8)
    const float LO = -6.2383246250395077f, HIc = -3.4657359027997265f;
    const float lin   = LO + (HIc - LO) * ((float)h / 7.0f);
    const float gamma = 1.0f - __expf(lin);
    const float lg    = __logf(gamma);

    const ushort_t* qrow  = qbf + ((size_t)(b * NH + h) * Ssz + n0 + c) * HD;
    const ushort_t* kbase = kbf + (size_t)(b * NH + h) * Ssz * HD;
    const ushort_t* vbase = vTbf + (size_t)(b * NH + h) * HD * Ssz;

    v16bf aq[4];
#pragma unroll
    for (int j = 0; j < 4; ++j) aq[j] = load_a16(qrow, 32 * j, hi);

    v8f racc[8];
#pragma unroll
    for (int dt = 0; dt < 8; ++dt) racc[dt] = (v8f){};

    for (int mc = 0; mc <= n0 + 15; mc += 32) {
        // prime next chunk's key rows in L2/WGP$ while we compute this one
        __builtin_prefetch(kbase + (size_t)(mc + 32 + c) * HD, 0, 1);
        __builtin_prefetch(kbase + (size_t)(mc + 48 + c) * HD, 0, 1);

        const ushort_t* k0row = kbase + (size_t)(mc + c) * HD;
        const ushort_t* k1row = kbase + (size_t)(mc + 16 + c) * HD;
        v16bf kb0[4], kb1[4];
#pragma unroll
        for (int j = 0; j < 4; ++j) {
            kb0[j] = load_b16(k0row, 32 * j, hi);
            kb1[j] = load_b16(k1row, 32 * j, hi);
        }
        v8f s0 = {}, s1 = {};
#pragma unroll
        for (int j = 0; j < 4; ++j) {
            s0 = wmma_bf(aq[j], kb0[j], s0);
            s1 = wmma_bf(aq[j], kb1[j], s1);
        }
        // decay mask + causal, then stage 16x32 bf16 score tile in LDS
#pragma unroll
        for (int g = 0; g < 8; ++g) {
            const int n  = n0 + g + 8 * hi;
            const int m0 = mc + c, m1 = mc + 16 + c;
            const float d0v = (n >= m0) ? __expf(lg * (float)(n - m0)) : 0.0f;
            const float d1v = (n >= m1) ? __expf(lg * (float)(n - m1)) : 0.0f;
            sc[wid][g + 8 * hi][c]      = f2bf(s0[g] * d0v);
            sc[wid][g + 8 * hi][16 + c] = f2bf(s1[g] * d1v);
        }
        __builtin_amdgcn_wave_barrier();
        asm volatile("s_wait_dscnt 0" ::: "memory");   // LDS stores visible (in-order DS)
        V16 sa;
        sa.q[0] = *(const u32x4*)&sc[wid][c][hi * 8];
        sa.q[1] = *(const u32x4*)&sc[wid][c][16 + hi * 8];
        __builtin_amdgcn_wave_barrier();

        v16bf vb[8];
#pragma unroll
        for (int dt = 0; dt < 8; ++dt)
            vb[dt] = load_b16(vbase + (size_t)(dt * 16 + c) * Ssz, mc, hi);
#pragma unroll
        for (int dt = 0; dt < 8; ++dt)
            racc[dt] = wmma_bf(sa.v, vb[dt], racc[dt]);
    }

    // GroupNorm over d=128 per row, + affine (gn_w, gn_b), write fp32 ret.
#pragma unroll
    for (int g = 0; g < 8; ++g) {
        float s = 0.0f, s2 = 0.0f;
#pragma unroll
        for (int dt = 0; dt < 8; ++dt) { const float v = racc[dt][g]; s += v; s2 += v * v; }
#pragma unroll
        for (int m = 1; m < 16; m <<= 1) {
            s  += __shfl_xor(s,  m, 32);
            s2 += __shfl_xor(s2, m, 32);
        }
        const float mean = s * (1.0f / 128.0f);
        const float var  = s2 * (1.0f / 128.0f) - mean * mean;
        const float rs   = rsqrtf(var + 1e-5f);
        const int   n    = n0 + g + 8 * hi;
        float* orow = ret + ((size_t)b * Ssz + n) * HID + h * HD;
#pragma unroll
        for (int dt = 0; dt < 8; ++dt) {
            const int col = dt * 16 + c;
            const int ch  = h * HD + col;
            orow[col] = (racc[dt][g] - mean) * rs * gnw[ch] + gnb[ch];
        }
    }
}

// ---------------------------------------------------------------------------
// K3: gated = x@W1 ; pre = swish(gated) + ret  (bf16 out)
//   One wave = 16 rows x 64 cols (4 tiles); stage A + 4 B per chunk.
// ---------------------------------------------------------------------------
__global__ __launch_bounds__(256) void gate_kernel(const ushort_t* __restrict__ xbf,
                                                   const ushort_t* __restrict__ w1T,
                                                   const float* __restrict__ ret,
                                                   ushort_t* __restrict__ prebf) {
    const int lane = threadIdx.x & 31;
    const int wid  = threadIdx.x >> 5;
    int t = blockIdx.x * 8 + wid;                     // 256 * 16 = 4096 tasks
    const int rb = t / 16, cg = t % 16;
    const int r0 = rb * 16, c0 = cg * 64;
    const int c = lane & 15, hi = lane >> 4;

    const ushort_t* arow = xbf + (size_t)(r0 + c) * HID;
    const ushort_t* bt0  = w1T + (size_t)(c0 + c) * HID;

    v8f acc[4];
#pragma unroll
    for (int ct = 0; ct < 4; ++ct) acc[ct] = (v8f){};

#pragma unroll 4
    for (int kc = 0; kc < HID; kc += 32) {
        __builtin_prefetch(arow + kc + 128, 0, 1);    // stream A ahead
        const v16bf a = load_a16(arow, kc, hi);
        v16bf bb[4];
#pragma unroll
        for (int ct = 0; ct < 4; ++ct)
            bb[ct] = load_b16(bt0 + (size_t)(ct * 16) * HID, kc, hi);
#pragma unroll
        for (int ct = 0; ct < 4; ++ct)
            acc[ct] = wmma_bf(a, bb[ct], acc[ct]);
    }

#pragma unroll
    for (int ct = 0; ct < 4; ++ct) {
#pragma unroll
        for (int g = 0; g < 8; ++g) {
            const int row = r0 + g + 8 * hi;
            const int col = c0 + ct * 16 + c;
            const float gg = acc[ct][g];
            const float sw = gg / (1.0f + __expf(-gg));   // swish
            prebf[(size_t)row * HID + col] = f2bf(sw + ret[(size_t)row * HID + col]);
        }
    }
}

// ---------------------------------------------------------------------------
// K4: out = pre @ W2  (fp32 out), same 16x64 macro-tile per wave.
// ---------------------------------------------------------------------------
__global__ __launch_bounds__(256) void out_kernel(const ushort_t* __restrict__ prebf,
                                                  const ushort_t* __restrict__ w2T,
                                                  float* __restrict__ out) {
    const int lane = threadIdx.x & 31;
    const int wid  = threadIdx.x >> 5;
    int t = blockIdx.x * 8 + wid;
    const int rb = t / 16, cg = t % 16;
    const int r0 = rb * 16, c0 = cg * 64;
    const int c = lane & 15, hi = lane >> 4;

    const ushort_t* arow = prebf + (size_t)(r0 + c) * HID;
    const ushort_t* bt0  = w2T + (size_t)(c0 + c) * HID;

    v8f acc[4];
#pragma unroll
    for (int ct = 0; ct < 4; ++ct) acc[ct] = (v8f){};

#pragma unroll 4
    for (int kc = 0; kc < HID; kc += 32) {
        __builtin_prefetch(arow + kc + 128, 0, 1);
        const v16bf a = load_a16(arow, kc, hi);
        v16bf bb[4];
#pragma unroll
        for (int ct = 0; ct < 4; ++ct)
            bb[ct] = load_b16(bt0 + (size_t)(ct * 16) * HID, kc, hi);
#pragma unroll
        for (int ct = 0; ct < 4; ++ct)
            acc[ct] = wmma_bf(a, bb[ct], acc[ct]);
    }

#pragma unroll
    for (int ct = 0; ct < 4; ++ct) {
#pragma unroll
        for (int g = 0; g < 8; ++g) {
            const int row = r0 + g + 8 * hi;
            out[(size_t)row * HID + c0 + ct * 16 + c] = acc[ct][g];
        }
    }
}

// ---------------------------------------------------------------------------
extern "C" void kernel_launch(void* const* d_in, const int* in_sizes, int n_in,
                              void* d_out, int out_size, void* d_ws, size_t ws_size,
                              hipStream_t stream) {
    const float* x   = (const float*)d_in[0];
    const float* Wq  = (const float*)d_in[1];
    const float* Wk  = (const float*)d_in[2];
    const float* Wv  = (const float*)d_in[3];
    const float* W1  = (const float*)d_in[4];
    const float* W2  = (const float*)d_in[5];
    const float* gnw = (const float*)d_in[6];
    const float* gnb = (const float*)d_in[7];
    float* out = (float*)d_out;

    // workspace layout (bf16 elements unless noted)
    ushort_t* ws = (ushort_t*)d_ws;
    ushort_t* xbf   = ws;                        // 4096*1024
    ushort_t* wT    = xbf   + (size_t)ROWS * HID;        // 3*8*128*128
    ushort_t* w1T   = wT    + (size_t)3 * NH * HD * HD;  // 1024*1024
    ushort_t* w2T   = w1T   + (size_t)HID * HID;
    ushort_t* qbf   = w2T   + (size_t)HID * HID;         // [B,H,S,128]
    ushort_t* kbf   = qbf   + (size_t)Bsz * NH * Ssz * HD;
    ushort_t* vTbf  = kbf   + (size_t)Bsz * NH * Ssz * HD;   // [B,H,128,S]
    ushort_t* prebf = vTbf  + (size_t)Bsz * NH * Ssz * HD;   // [rows,1024]
    float*    retb  = (float*)(prebf + (size_t)ROWS * HID);  // fp32 [rows,1024]

    // conversions
    cvt_bf16_kernel<<<(ROWS * HID) / 256, 256, 0, stream>>>(x, xbf, ROWS * HID);
    transpose_head_kernel<<<(NH * HD * HD) / 256, 256, 0, stream>>>(Wq, wT);
    transpose_head_kernel<<<(NH * HD * HD) / 256, 256, 0, stream>>>(Wk, wT + NH * HD * HD);
    transpose_head_kernel<<<(NH * HD * HD) / 256, 256, 0, stream>>>(Wv, wT + 2 * NH * HD * HD);
    transpose_1024_kernel<<<(HID * HID) / 256, 256, 0, stream>>>(W1, w1T);
    transpose_1024_kernel<<<(HID * HID) / 256, 256, 0, stream>>>(W2, w2T);

    // QKV projections: 3 * 2*8*128 = 6144 wave row-blocks / 8 per block
    qkv_kernel<<<768, 256, 0, stream>>>(xbf, wT, qbf, kbf, vTbf);

    // retention + groupnorm: 2048 wave tasks / 8 per block
    attn_kernel<<<256, 256, 0, stream>>>(qbf, kbf, vTbf, gnw, gnb, retb);

    // swish gate + residual: 4096 macro-tiles / 8
    gate_kernel<<<512, 256, 0, stream>>>(xbf, w1T, retb, prebf);

    // final projection: 4096 macro-tiles / 8
    out_kernel<<<512, 256, 0, stream>>>(prebf, w2T, out);
}
